// _FwdAttention_3599182594192
// MI455X (gfx1250) — compile-verified, matrix operand order checked by reading, not run on hardware
//
#include <hip/hip_runtime.h>

typedef unsigned int  u32;
typedef unsigned short u16;

#define NB   32      // batches
#define SS   4096    // sequence
#define CD   512     // CTXD
#define QDIM 512     // QD
#define AD   256     // ATTD

#define ROWS 64            // ctx rows per block in scores kernel
#define NBLK (SS / ROWS)   // 64 score-blocks per batch
#define STRW 257           // LDS dwords per row (514 bf16, odd -> conflict-free frag loads)

typedef __attribute__((ext_vector_type(16))) __bf16 v16bf;
typedef __attribute__((ext_vector_type(8)))  float  v8f;

__device__ __forceinline__ u32 pack_bf16(float a, float b) {
  u32 ua = __float_as_uint(a), ub = __float_as_uint(b);
  ua += 0x7FFFu + ((ua >> 16) & 1u);            // round-to-nearest-even
  ub += 0x7FFFu + ((ub >> 16) & 1u);
  return (ua >> 16) | (ub & 0xFFFF0000u);
}

// ---------------------------------------------------------------------------
// Kernel 0: swizzle W_ctx (512x256 f32, row-major) -> bf16 fragments in the
// ISA B-matrix layout: for tile (kt,nt), lane l holds column n=nt*16+(l&15);
// halves j=2v+p hold k = kt*32 + ((v>=4)?16:0) + (l>>4)*8 + (v&3)*2 + p.
// Stored lane-contiguous: frag(kt,nt) base = ((kt*16+nt)*32 + lane)*16 halves.
// ---------------------------------------------------------------------------
__global__ void __launch_bounds__(32) prep_w(const float* __restrict__ W,
                                             u16* __restrict__ Wsw) {
  int tile = blockIdx.x;           // 0..255
  int kt = tile >> 4;              // 16 k-tiles (K=512 / 32)
  int nt = tile & 15;              // 16 n-tiles (N=256 / 16)
  int lane = threadIdx.x;          // 0..31
  int n = nt * 16 + (lane & 15);
  int g = lane >> 4;
  u32 out[8];
#pragma unroll
  for (int v = 0; v < 8; ++v) {
    int k = kt * 32 + ((v >= 4) ? 16 : 0) + g * 8 + (v & 3) * 2;
    float f0 = W[(size_t)k * AD + n];
    float f1 = W[(size_t)(k + 1) * AD + n];
    out[v] = pack_bf16(f0, f1);
  }
  u32* dst = (u32*)(Wsw + ((size_t)((kt * 16 + nt) * 32 + lane) * 16));
#pragma unroll
  for (int v = 0; v < 8; ++v) dst[v] = out[v];
}

// ---------------------------------------------------------------------------
// Kernel 1: qproj[b][n] = qry[b,:] @ W_qry[:,n] + bias[n]   (f32, tiny)
// ---------------------------------------------------------------------------
__global__ void __launch_bounds__(256) prep_q(const float* __restrict__ qry,
                                              const float* __restrict__ W,
                                              const float* __restrict__ bias,
                                              float* __restrict__ qproj) {
  int b = blockIdx.x, n = threadIdx.x;
  const float* q  = qry + (size_t)b * QDIM;
  const float* Wq = W + (size_t)CD * AD;
  float acc = bias[n];
  for (int k = 0; k < QDIM; ++k) acc += q[k] * Wq[(size_t)k * AD + n];
  qproj[(size_t)b * AD + n] = acc;
}

// ---------------------------------------------------------------------------
// Kernel 2: fused  h = ctx@W_ctx + qproj;  s = tanh(h)@w2 + b2 + log(mask);
// plus per-block UNNORMALIZED summary partial:  psum = sum_m exp(s_m)*ctx[m,:]
// (scores bounded by |w2|_1 ~ 16 => exp(s) < 1e7, safe in f32 without a max).
// Grid (SS/ROWS, NB), 256 threads = 8 waves. Wave w: row-tile rt=w>>1,
// column half ch=w&1 (8 n-tiles). K streamed in 16 steps of 32 (bf16 WMMA).
// ---------------------------------------------------------------------------
__global__ void __launch_bounds__(256) attn_scores(
    const float* __restrict__ ctx, const int* __restrict__ mask,
    const u16* __restrict__ Wsw, const float* __restrict__ qproj,
    const float* __restrict__ w2, const float* __restrict__ b2,
    float* __restrict__ scores_out, float* __restrict__ psum) {
  __shared__ u32 ldsA[ROWS * STRW];     // 64 rows x 514 bf16 (padded)
  __shared__ float ldsS[2][ROWS];       // per-column-half score partials
  __shared__ float ldsW[ROWS];          // exp-weights for fused summary

  const int t = threadIdx.x;
  const int batch = blockIdx.y;
  const int row0 = blockIdx.x * ROWS;
  const float* src = ctx + ((size_t)batch * SS + row0) * CD;

  // ---- stage ctx tile -> bf16 in LDS (coalesced float4 loads) ----
#pragma unroll
  for (int i = 0; i < 32; ++i) {
    int flat = i * 256 + t;             // 8192 float4 total
    int r  = flat >> 7;                 // 128 float4 per row
    int c4 = flat & 127;
    float4 f = ((const float4*)src)[(size_t)r * 128 + c4];
    ldsA[r * STRW + c4 * 2]     = pack_bf16(f.x, f.y);
    ldsA[r * STRW + c4 * 2 + 1] = pack_bf16(f.z, f.w);
  }
  __syncthreads();

  const int wave = t >> 5, lane = t & 31;
  const int rt = wave >> 1;             // 0..3
  const int ch = wave & 1;              // 0..1
  const int r  = lane & 15;
  const int g  = lane >> 4;

  v8f acc[8] = {};
  const u32 rowBase = (u32)((rt * 16 + r) * STRW);

  for (int kt = 0; kt < 16; ++kt) {
    // A fragment (16x32 bf16) from LDS, ISA 16-bit A layout
    union { v16bf v; u32 u[8]; } A;
    const u32 kw = (u32)(kt * 16 + g * 4);
#pragma unroll
    for (int v = 0; v < 8; ++v)
      A.u[v] = ldsA[rowBase + kw + (u32)(((v >> 2) << 3) + (v & 3))];

    const u16* wb = Wsw + ((size_t)((kt * 16 + ch * 8) * 32 + lane) * 16);
#pragma unroll
    for (int i = 0; i < 8; ++i) {
      v16bf Bf = *(const v16bf*)(wb + (size_t)i * 512);   // 32*16 halves per tile
      acc[i] = __builtin_amdgcn_wmma_f32_16x16x32_bf16(
          false, A.v, false, Bf, (short)0, acc[i], false, false);
    }
  }

  // ---- epilogue: +qproj, tanh, *w2, reduce over N ----
  float part[8];
#pragma unroll
  for (int v = 0; v < 8; ++v) part[v] = 0.f;
#pragma unroll
  for (int i = 0; i < 8; ++i) {
    int n = (ch * 8 + i) * 16 + r;
    float qp = qproj[(size_t)batch * AD + n];
    float wn = w2[n];
#pragma unroll
    for (int v = 0; v < 8; ++v) {
      float h = acc[i][v] + qp;
      float e = __expf(2.f * h);
      part[v] += (1.f - 2.f / (e + 1.f)) * wn;   // tanh(h) * w2[n]
    }
  }
  // reduce across the 16 lanes of each half (C/D layout: N = lane&15)
#pragma unroll
  for (int m = 1; m <= 8; m <<= 1)
#pragma unroll
    for (int v = 0; v < 8; ++v) part[v] += __shfl_xor(part[v], m, 32);
  if (r == 0) {
#pragma unroll
    for (int v = 0; v < 8; ++v) ldsS[ch][rt * 16 + g * 8 + v] = part[v];
  }
  __syncthreads();

  if (t < ROWS) {
    float s = ldsS[0][t] + ldsS[1][t] + b2[0];
    int mk = mask[(size_t)batch * SS + row0 + t];
    scores_out[(size_t)batch * SS + row0 + t] = mk ? s : -__builtin_inff();
    ldsW[t] = mk ? __expf(s) : 0.f;              // unnormalized softmax weight
  }
  __syncthreads();

  // ---- fused summary partial: re-read the f32 ctx tile (L2-hot) ----
  {
    float a0 = 0.f, a1 = 0.f;
    const int c0 = t, c1 = t + 256;
    for (int m = 0; m < ROWS; ++m) {
      float wt = ldsW[m];
      if (wt != 0.f) {                           // wave-uniform: skip masked rows
        a0 += src[(size_t)m * CD + c0] * wt;
        a1 += src[(size_t)m * CD + c1] * wt;
      }
    }
    float* p = psum + ((size_t)batch * NBLK + blockIdx.x) * CD;
    p[c0] = a0; p[c1] = a1;
  }
}

// ---------------------------------------------------------------------------
// Kernel 3: per-batch softmax over S (deterministic tree reductions).
// Also emits 1 / sum_i exp(s_i)  (unnormalized denom) for the summary scale.
// ---------------------------------------------------------------------------
__global__ void __launch_bounds__(256) softmax_k(const float* __restrict__ scores,
                                                 float* __restrict__ alphas,
                                                 float* __restrict__ invdenom) {
  __shared__ float red[256];
  int b = blockIdx.x, t = threadIdx.x;
  const float* s = scores + (size_t)b * SS;
  float* a = alphas + (size_t)b * SS;

  float mx = -__builtin_inff();
  for (int i = t; i < SS; i += 256) mx = fmaxf(mx, s[i]);
  red[t] = mx; __syncthreads();
  for (int o = 128; o > 0; o >>= 1) {
    if (t < o) red[t] = fmaxf(red[t], red[t + o]);
    __syncthreads();
  }
  mx = red[0]; __syncthreads();

  float sum = 0.f;
  for (int i = t; i < SS; i += 256) { float e = __expf(s[i] - mx); a[i] = e; sum += e; }
  red[t] = sum; __syncthreads();
  for (int o = 128; o > 0; o >>= 1) {
    if (t < o) red[t] += red[t + o];
    __syncthreads();
  }
  float inv = 1.f / red[0];
  for (int i = t; i < SS; i += 256) a[i] *= inv;
  if (t == 0) invdenom[b] = 1.f / (__expf(mx) * red[0]); // = 1 / sum exp(s)
}

// ---------------------------------------------------------------------------
// Kernel 4: summary[b][c] = (sum over 64 block partials) * invdenom[b]
// Fixed-order reduction over 4 MB of L2-resident partials -> deterministic.
// ---------------------------------------------------------------------------
__global__ void __launch_bounds__(256) summary_final(const float* __restrict__ psum,
                                                     const float* __restrict__ invdenom,
                                                     float* __restrict__ summary) {
  int b = blockIdx.x, t = threadIdx.x;
  float a0 = 0.f, a1 = 0.f;
  for (int c = 0; c < NBLK; ++c) {
    const float* p = psum + ((size_t)b * NBLK + c) * CD;
    a0 += p[t]; a1 += p[t + 256];
  }
  float inv = invdenom[b];
  summary[(size_t)b * CD + t]       = a0 * inv;
  summary[(size_t)b * CD + t + 256] = a1 * inv;
}

// ---------------------------------------------------------------------------
extern "C" void kernel_launch(void* const* d_in, const int* in_sizes, int n_in,
                              void* d_out, int out_size, void* d_ws, size_t ws_size,
                              hipStream_t stream) {
  const float* qry  = (const float*)d_in[0];
  const float* ctx  = (const float*)d_in[1];
  const int*   mask = (const int*)  d_in[2];
  const float* W    = (const float*)d_in[3];
  const float* bias = (const float*)d_in[4];
  const float* w2   = (const float*)d_in[5];
  const float* b2   = (const float*)d_in[6];

  float* out     = (float*)d_out;
  float* alphas  = out;                                     // B*S
  float* summary = out + (size_t)NB * SS;                   // B*CD
  float* scores  = out + (size_t)NB * SS + (size_t)NB * CD; // B*S

  char* ws = (char*)d_ws;
  u16*   Wsw   = (u16*)ws;                                  // 256 KB swizzled bf16 W_ctx
  float* qproj = (float*)(ws + 262144);                     // 32 KB
  float* psum  = (float*)(ws + 262144 + 32768);             // 4 MB block partials
  float* invd  = (float*)(ws + 262144 + 32768 + (size_t)NB * NBLK * CD * 4); // 128 B

  prep_w<<<256, 32, 0, stream>>>(W, Wsw);
  prep_q<<<NB, 256, 0, stream>>>(qry, W, bias, qproj);
  attn_scores<<<dim3(NBLK, NB), 256, 0, stream>>>(ctx, mask, Wsw, qproj, w2, b2,
                                                  scores, psum);
  softmax_k<<<NB, 256, 0, stream>>>(scores, alphas, invd);
  summary_final<<<NB, 256, 0, stream>>>(psum, invd, summary);
}
